// MultiSimilarityLoss_53764400611385
// MI455X (gfx1250) — matrix operand order repaired and checked by reading, not compiled, not run on hardware
//
#include <hip/hip_runtime.h>
#include <cstdint>
#include <cstddef>

#define B_N  8192
#define D_K  128
#define NB   512     // number of 16-wide row/col blocks
#define KT   4       // number of K=32 tiles covering D=128
#define BIGF 3.0e38f

typedef __attribute__((ext_vector_type(16))) __bf16          v16bf;
typedef __attribute__((ext_vector_type(16))) unsigned short  v16us;
typedef __attribute__((ext_vector_type(8)))  float           v8f;

__device__ __forceinline__ unsigned short f32_to_bf16_rne(float f) {
    unsigned int x = __float_as_uint(f);
    unsigned int r = x + 0x7FFFu + ((x >> 16) & 1u);
    return (unsigned short)(r >> 16);
}
__device__ __forceinline__ float bf16_bits_to_f32(unsigned short h) {
    return __uint_as_float(((unsigned int)h) << 16);
}

// ---------------------------------------------------------------------------
// K0: split f32 embeddings into (hi, lo) bf16 and pre-pack into WMMA fragment
// lane layouts (A: 16x32 bf16 layout; B: 32x16 bf16 layout) so the GEMM loop
// does pure contiguous 32B loads. 1 thread per packed element.
// ---------------------------------------------------------------------------
__global__ void pack_kernel(const float* __restrict__ emb,
                            unsigned short* __restrict__ pAhi,
                            unsigned short* __restrict__ pAlo,
                            unsigned short* __restrict__ pBhi,
                            unsigned short* __restrict__ pBlo) {
    int tid  = blockIdx.x * blockDim.x + threadIdx.x;   // < B_N * D_K
    int e    = tid & 15;
    int lane = (tid >> 4) & 31;
    int kt   = (tid >> 9) & 3;
    int blk  = tid >> 11;
    int hb   = lane >> 4;                    // half-wave select
    int row  = blk * 16 + (lane & 15);
    // B fragment (32x16, lane = column): lanes 0-15 hold K=0..15, 16-31 hold K=16..31
    int kB = kt * 32 + e + (hb << 4);
    // A fragment (16x32, lane%16 = row): lanes 0-15 hold K=0..15, 16-31 hold K=16..31
    int kA = kt * 32 + e + (hb << 4);
    float fB = emb[row * D_K + kB];
    float fA = emb[row * D_K + kA];
    unsigned short hB = f32_to_bf16_rne(fB);
    unsigned short hA = f32_to_bf16_rne(fA);
    pBhi[tid] = hB;
    pAhi[tid] = hA;
    pBlo[tid] = f32_to_bf16_rne(fB - bf16_bits_to_f32(hB));
    pAlo[tid] = f32_to_bf16_rne(fA - bf16_bits_to_f32(hA));
}

__device__ __forceinline__ v16bf load_frag(const unsigned short* __restrict__ base,
                                           int blk, int kt, int lane) {
    const v16us* p = (const v16us*)base;
    v16us t = p[(blk * 4 + kt) * 32 + lane];
    return __builtin_bit_cast(v16bf, t);
}

struct BTile { v16bf bh[KT]; v16bf bl[KT]; };

__device__ __forceinline__ BTile load_tile(const unsigned short* __restrict__ pBhi,
                                           const unsigned short* __restrict__ pBlo,
                                           int jb, int lane) {
    BTile t;
#pragma unroll
    for (int kt = 0; kt < KT; ++kt) {
        t.bh[kt] = load_frag(pBhi, jb, kt, lane);
        t.bl[kt] = load_frag(pBlo, jb, kt, lane);
    }
    return t;
}

// 16x16 fp32 sim tile via error-compensated bf16 WMMA:
//   A*B ~= Ahi*Blo + Alo*Bhi + Ahi*Bhi  (lo*lo term ~2^-16 relative, dropped)
__device__ __forceinline__ v8f compute_tile(const v16bf aHi[KT], const v16bf aLo[KT],
                                            const BTile& t) {
    v8f acc = {0.f, 0.f, 0.f, 0.f, 0.f, 0.f, 0.f, 0.f};
#pragma unroll
    for (int kt = 0; kt < KT; ++kt) {
        acc = __builtin_amdgcn_wmma_f32_16x16x32_bf16(false, aHi[kt], false, t.bl[kt],
                                                      (short)0, acc, false, false);
        acc = __builtin_amdgcn_wmma_f32_16x16x32_bf16(false, aLo[kt], false, t.bh[kt],
                                                      (short)0, acc, false, false);
        acc = __builtin_amdgcn_wmma_f32_16x16x32_bf16(false, aHi[kt], false, t.bh[kt],
                                                      (short)0, acc, false, false);
    }
    return acc;
}

// ---------------------------------------------------------------------------
// Pass 1: per-row mining thresholds neg_max / pos_min (sentinels +-3e38) plus
// the threshold-independent full-mask exp sums spa / sna.
// One block per 16-row stripe; 8 waves software-pipeline over column tiles:
// loads for tile i+1 are pinned above tile i's WMMAs with sched_barrier(0).
// ---------------------------------------------------------------------------
__global__ void __launch_bounds__(256)
pass1_kernel(const unsigned short* __restrict__ pAhi, const unsigned short* __restrict__ pAlo,
             const unsigned short* __restrict__ pBhi, const unsigned short* __restrict__ pBlo,
             const int* __restrict__ labels, const float* __restrict__ ages,
             const float* __restrict__ posThrP, const float* __restrict__ negThrP,
             float* __restrict__ negMaxOut, float* __restrict__ posMinOut,
             float* __restrict__ spaOut, float* __restrict__ snaOut) {
    const int ib   = blockIdx.x;
    const int lane = threadIdx.x & 31;
    const int wave = threadIdx.x >> 5;
    const int hb   = lane >> 4;
    const float pT = posThrP[0], nT = negThrP[0];

    v16bf aHi[KT], aLo[KT];
#pragma unroll
    for (int kt = 0; kt < KT; ++kt) {
        aHi[kt] = load_frag(pAhi, ib, kt, lane);
        aLo[kt] = load_frag(pAlo, ib, kt, lane);
    }

    const int rowBase = ib * 16 + hb * 8;   // D-fragment: VGPR r -> row rowBase + r
    int   labR[8]; float ageR[8];
#pragma unroll
    for (int r = 0; r < 8; ++r) { labR[r] = labels[rowBase + r]; ageR[r] = ages[rowBase + r]; }

    float nmx[8], pmn[8], spa[8], sna[8];
#pragma unroll
    for (int r = 0; r < 8; ++r) { nmx[r] = -BIGF; pmn[r] = BIGF; spa[r] = 0.f; sna[r] = 0.f; }

    BTile cur = load_tile(pBhi, pBlo, wave, lane);
    for (int jb = wave; jb < NB; jb += 8) {
        int jn = (jb + 8 < NB) ? jb + 8 : jb;
        BTile nxt = load_tile(pBhi, pBlo, jn, lane);     // prefetch next tile
        __builtin_amdgcn_sched_barrier(0);               // keep loads above WMMAs
        int   col  = jb * 16 + (lane & 15);
        int   labC = labels[col];
        float ageC = ages[col];
        v8f s = compute_tile(aHi, aLo, cur);
#pragma unroll
        for (int r = 0; r < 8; ++r) {
            int   row  = rowBase + r;
            bool  offd = (row != col);
            bool  same = (labR[r] == labC);
            float ad   = fabsf(ageR[r] - ageC);
            bool  pos  = same && (ad <= pT) && offd;
            bool  neg  = ((!same) || (ad >= nT)) && offd;
            float sv   = s[r];
            nmx[r] = fmaxf(nmx[r], neg ? sv : -BIGF);
            pmn[r] = fminf(pmn[r], pos ? sv :  BIGF);
            float e1 = __expf(fminf(-2.0f * (sv - 0.5f), 50.0f));   // alpha=2, base=0.5
            float e2 = __expf(fminf(50.0f * (sv - 0.5f), 50.0f));   // beta=50
            spa[r] += pos ? e1 : 0.f;
            sna[r] += neg ? e2 : 0.f;
        }
        cur = nxt;
    }
    // reduce across the 16 lanes of each half-wave (xor masks stay within halves)
#pragma unroll
    for (int r = 0; r < 8; ++r) {
#pragma unroll
        for (int m = 1; m < 16; m <<= 1) {
            nmx[r] = fmaxf(nmx[r], __shfl_xor(nmx[r], m, 32));
            pmn[r] = fminf(pmn[r], __shfl_xor(pmn[r], m, 32));
            spa[r] += __shfl_xor(spa[r], m, 32);
            sna[r] += __shfl_xor(sna[r], m, 32);
        }
    }
    __shared__ float sh[4][8][16];
    if ((lane & 15) == 0) {
#pragma unroll
        for (int r = 0; r < 8; ++r) {
            sh[0][wave][hb * 8 + r] = nmx[r];
            sh[1][wave][hb * 8 + r] = pmn[r];
            sh[2][wave][hb * 8 + r] = spa[r];
            sh[3][wave][hb * 8 + r] = sna[r];
        }
    }
    __syncthreads();
    if (threadIdx.x < 16) {
        float mn = -BIGF, mp = BIGF, a = 0.f, b = 0.f;
#pragma unroll
        for (int w = 0; w < 8; ++w) {
            mn = fmaxf(mn, sh[0][w][threadIdx.x]);
            mp = fminf(mp, sh[1][w][threadIdx.x]);
            a += sh[2][w][threadIdx.x];
            b += sh[3][w][threadIdx.x];
        }
        negMaxOut[ib * 16 + threadIdx.x] = mn;
        posMinOut[ib * 16 + threadIdx.x] = mp;
        spaOut[ib * 16 + threadIdx.x]    = a;
        snaOut[ib * 16 + threadIdx.x]    = b;
    }
}

// ---------------------------------------------------------------------------
// Pass 2: recompute sim tiles, accumulate per-row keep-set exp sums.
// sum_keep > 0  <=>  any kept (exp terms are strictly positive).
// ---------------------------------------------------------------------------
__global__ void __launch_bounds__(256)
pass2_kernel(const unsigned short* __restrict__ pAhi, const unsigned short* __restrict__ pAlo,
             const unsigned short* __restrict__ pBhi, const unsigned short* __restrict__ pBlo,
             const int* __restrict__ labels, const float* __restrict__ ages,
             const float* __restrict__ posThrP, const float* __restrict__ negThrP,
             const float* __restrict__ negMax, const float* __restrict__ posMin,
             float* __restrict__ spkOut, float* __restrict__ snkOut) {
    const int ib   = blockIdx.x;
    const int lane = threadIdx.x & 31;
    const int wave = threadIdx.x >> 5;
    const int hb   = lane >> 4;
    const float pT = posThrP[0], nT = negThrP[0];

    v16bf aHi[KT], aLo[KT];
#pragma unroll
    for (int kt = 0; kt < KT; ++kt) {
        aHi[kt] = load_frag(pAhi, ib, kt, lane);
        aLo[kt] = load_frag(pAlo, ib, kt, lane);
    }

    const int rowBase = ib * 16 + hb * 8;
    int   labR[8]; float ageR[8], nmR[8], pmR[8];
#pragma unroll
    for (int r = 0; r < 8; ++r) {
        labR[r] = labels[rowBase + r];
        ageR[r] = ages[rowBase + r];
        nmR[r]  = negMax[rowBase + r] + 0.1f;   // fold in margin
        pmR[r]  = posMin[rowBase + r] - 0.1f;
    }

    float spk[8], snk[8];
#pragma unroll
    for (int r = 0; r < 8; ++r) { spk[r] = 0.f; snk[r] = 0.f; }

    BTile cur = load_tile(pBhi, pBlo, wave, lane);
    for (int jb = wave; jb < NB; jb += 8) {
        int jn = (jb + 8 < NB) ? jb + 8 : jb;
        BTile nxt = load_tile(pBhi, pBlo, jn, lane);     // prefetch next tile
        __builtin_amdgcn_sched_barrier(0);               // keep loads above WMMAs
        int   col  = jb * 16 + (lane & 15);
        int   labC = labels[col];
        float ageC = ages[col];
        v8f s = compute_tile(aHi, aLo, cur);
#pragma unroll
        for (int r = 0; r < 8; ++r) {
            int   row  = rowBase + r;
            bool  offd = (row != col);
            bool  same = (labR[r] == labC);
            float ad   = fabsf(ageR[r] - ageC);
            bool  pos  = same && (ad <= pT) && offd;
            bool  neg  = ((!same) || (ad >= nT)) && offd;
            float sv   = s[r];
            float e1 = __expf(fminf(-2.0f * (sv - 0.5f), 50.0f));   // alpha=2, base=0.5
            float e2 = __expf(fminf(50.0f * (sv - 0.5f), 50.0f));   // beta=50
            bool  pk = pos && (sv < nmR[r]);
            bool  nk = neg && (sv > pmR[r]);
            spk[r] += pk ? e1 : 0.f;
            snk[r] += nk ? e2 : 0.f;
        }
        cur = nxt;
    }
#pragma unroll
    for (int r = 0; r < 8; ++r) {
#pragma unroll
        for (int m = 1; m < 16; m <<= 1) {
            spk[r] += __shfl_xor(spk[r], m, 32);
            snk[r] += __shfl_xor(snk[r], m, 32);
        }
    }
    __shared__ float sh[2][8][16];
    if ((lane & 15) == 0) {
#pragma unroll
        for (int r = 0; r < 8; ++r) {
            sh[0][wave][hb * 8 + r] = spk[r];
            sh[1][wave][hb * 8 + r] = snk[r];
        }
    }
    __syncthreads();
    if (threadIdx.x < 16) {
        float a = 0.f, c = 0.f;
#pragma unroll
        for (int w = 0; w < 8; ++w) {
            a += sh[0][w][threadIdx.x];
            c += sh[1][w][threadIdx.x];
        }
        spkOut[ib * 16 + threadIdx.x] = a;
        snkOut[ib * 16 + threadIdx.x] = c;
    }
}

// ---------------------------------------------------------------------------
// Final scalar reduction over 8192 rows.
// ---------------------------------------------------------------------------
__global__ void __launch_bounds__(256)
final_kernel(const float* __restrict__ negMax, const float* __restrict__ posMin,
             const float* __restrict__ spk, const float* __restrict__ spa,
             const float* __restrict__ snk, const float* __restrict__ sna,
             float* __restrict__ out) {
    __shared__ float sTot[256], sCnt[256];
    __shared__ int   sGP[256], sGN[256];
    float tot = 0.f, cnt = 0.f; int gp = 0, gn = 0;
    for (int i = threadIdx.x; i < B_N; i += 256) {
        bool posAny = posMin[i] <  2.0e38f;   // pos_min sentinel is +3e38
        bool negAny = negMax[i] > -2.0e38f;   // neg_max sentinel is -3e38
        float a = spk[i], b = spa[i], c = snk[i], d = sna[i];
        float sp = (a > 0.f) ? a : b;         // mining fallback
        float sn = (c > 0.f) ? c : d;
        float lr = 0.5f * log1pf(sp) + 0.02f * log1pf(sn);  // 1/alpha, 1/beta
        bool valid = posAny && negAny;
        tot += valid ? lr : 0.f;
        cnt += valid ? 1.f : 0.f;
        gp |= (int)posAny; gn |= (int)negAny;
    }
    sTot[threadIdx.x] = tot; sCnt[threadIdx.x] = cnt;
    sGP[threadIdx.x] = gp;   sGN[threadIdx.x] = gn;
    __syncthreads();
    for (int s = 128; s > 0; s >>= 1) {
        if (threadIdx.x < (unsigned)s) {
            sTot[threadIdx.x] += sTot[threadIdx.x + s];
            sCnt[threadIdx.x] += sCnt[threadIdx.x + s];
            sGP[threadIdx.x]  |= sGP[threadIdx.x + s];
            sGN[threadIdx.x]  |= sGN[threadIdx.x + s];
        }
        __syncthreads();
    }
    if (threadIdx.x == 0) {
        float loss = (sGP[0] && sGN[0]) ? sTot[0] / fmaxf(sCnt[0], 1.0f) : 0.0f;
        out[0] = loss;
    }
}

extern "C" void kernel_launch(void* const* d_in, const int* in_sizes, int n_in,
                              void* d_out, int out_size, void* d_ws, size_t ws_size,
                              hipStream_t stream) {
    const float* emb    = (const float*)d_in[0];
    const int*   labels = (const int*)  d_in[1];
    const float* ages   = (const float*)d_in[2];
    const float* posT   = (const float*)d_in[3];
    const float* negT   = (const float*)d_in[4];
    float* out = (float*)d_out;

    // Workspace layout (8 MB pack + 192 KB stats)
    char* w = (char*)d_ws;
    const size_t PK = (size_t)B_N * D_K;                 // 1,048,576 bf16 per array
    unsigned short* pAhi = (unsigned short*)(w + 0 * PK);
    unsigned short* pAlo = (unsigned short*)(w + 2 * PK);
    unsigned short* pBhi = (unsigned short*)(w + 4 * PK);
    unsigned short* pBlo = (unsigned short*)(w + 6 * PK);
    float* negMax = (float*)(w + 8 * PK);
    float* posMin = negMax + B_N;
    float* spk    = posMin + B_N;
    float* spa    = spk + B_N;
    float* snk    = spa + B_N;
    float* sna    = snk + B_N;

    pack_kernel<<<(int)(PK / 256), 256, 0, stream>>>(emb, pAhi, pAlo, pBhi, pBlo);
    pass1_kernel<<<NB, 256, 0, stream>>>(pAhi, pAlo, pBhi, pBlo, labels, ages,
                                         posT, negT, negMax, posMin, spa, sna);
    pass2_kernel<<<NB, 256, 0, stream>>>(pAhi, pAlo, pBhi, pBlo, labels, ages,
                                         posT, negT, negMax, posMin, spk, snk);
    final_kernel<<<1, 256, 0, stream>>>(negMax, posMin, spk, spa, snk, sna, out);
}
